// MultiHeadAttention_69922067579127
// MI455X (gfx1250) — compile-verified, matrix operand order checked by reading, not run on hardware
//
#include <hip/hip_runtime.h>

// ---------------- problem constants ----------------
static constexpr int BB   = 4;
static constexpr int T    = 2048;
static constexpr int C    = 1024;
static constexpr int H    = 16;
static constexpr int Dh   = 64;          // C / H
static constexpr int MTOT = BB * T;      // 8192 rows
static constexpr int N_QKV = 3 * C;      // 3072

// ---------------- workspace layout (bytes) ----------------
static constexpr size_t OFF_XB    = 0;                                    // x bf16      16 MB
static constexpr size_t OFF_WQT   = OFF_XB  + (size_t)MTOT * C * 2;       // Wqkv^T bf16  6 MB
static constexpr size_t OFF_WOT   = OFF_WQT + (size_t)N_QKV * C * 2;      // Wout^T bf16  2 MB
static constexpr size_t OFF_QB    = OFF_WOT + (size_t)C * C * 2;          // Q bf16      16 MB
static constexpr size_t OFF_KB    = OFF_QB  + (size_t)MTOT * C * 2;       // K bf16      16 MB
static constexpr size_t OFF_VB    = OFF_KB  + (size_t)MTOT * C * 2;       // V^T bf16    16 MB
static constexpr size_t OFF_ATT   = OFF_VB  + (size_t)MTOT * C * 2;       // attn bf16   16 MB

// ---------------- WMMA types ----------------
typedef __attribute__((ext_vector_type(16))) __bf16 v16bf;
typedef __attribute__((ext_vector_type(8)))  __bf16 v8bf;
typedef __attribute__((ext_vector_type(4)))  __bf16 v4bf;
typedef __attribute__((ext_vector_type(8)))  float  v8f;

__device__ __forceinline__ v8f wmma_bf16(v16bf a, v16bf b, v8f c) {
  // D = A(16x32) * B(32x16) + C, f32 accumulate
  return __builtin_amdgcn_wmma_f32_16x16x32_bf16(
      /*neg_a=*/false, a, /*neg_b=*/false, b,
      /*c_mod=*/(short)0, c, /*reuse_a=*/false, /*reuse_b=*/false);
}

// Load one 16x32 bf16 fragment (A-operand pattern; identical pattern serves the
// B operand when the source is stored as B^T, i.e. [N x K] row-major).
// lane group 0 (0-15): cols {0..7, 16..23}; group 1 (16-31): cols {8..15, 24..31}.
__device__ __forceinline__ v16bf load_frag_row(const __bf16* p, int ld) {
  const int lane = threadIdx.x & 31;
  const int row  = lane & 15;
  const int hi   = lane >> 4;
  const __bf16* b = p + (size_t)row * ld + hi * 8;
  union { v16bf v; v8bf h[2]; } u;
  u.h[0] = *(const v8bf*)(b);        // 16-byte load
  u.h[1] = *(const v8bf*)(b + 16);   // 16-byte load
  return u.v;
}

// ---------------- conversion / transpose kernels ----------------
__global__ void k_cvt_bf16(const float* __restrict__ src, __bf16* __restrict__ dst, int n4) {
  int i = blockIdx.x * blockDim.x + threadIdx.x;
  if (i >= n4) return;
  const float4 f = *(const float4*)(src + (size_t)i * 4);
  v4bf o;
  o[0] = (__bf16)f.x; o[1] = (__bf16)f.y; o[2] = (__bf16)f.z; o[3] = (__bf16)f.w;
  *(v4bf*)(dst + (size_t)i * 4) = o;
}

// dst[n*K + k] = (bf16)src[k*N + n] via 32x32 LDS tile (coalesced both sides)
__global__ __launch_bounds__(256) void k_transpose_bf16(
    const float* __restrict__ src, __bf16* __restrict__ dst, int K, int N) {
  __shared__ float tile[32][33];
  const int ntiles_n = N >> 5;
  const int bx = blockIdx.x % ntiles_n;   // n tile
  const int by = blockIdx.x / ntiles_n;   // k tile
  const int tx = threadIdx.x & 31;
  const int ty = threadIdx.x >> 5;        // 0..7
  const int n0 = bx * 32, k0 = by * 32;
#pragma unroll
  for (int r = 0; r < 4; ++r)
    tile[ty + r * 8][tx] = src[(size_t)(k0 + ty + r * 8) * N + n0 + tx];
  __syncthreads();
#pragma unroll
  for (int r = 0; r < 4; ++r)
    dst[(size_t)(n0 + ty + r * 8) * K + k0 + tx] = (__bf16)tile[tx][ty + r * 8];
}

// ---------------- shared GEMM machinery: 32(M) x 64(N) tile per wave ----------------
__device__ __forceinline__ void load_ab(const __bf16* Arow, const __bf16* Brow, int ks,
                                        v16bf (&a)[2], v16bf (&b)[4]) {
#pragma unroll
  for (int i = 0; i < 2; ++i) a[i] = load_frag_row(Arow + (size_t)i * 16 * C + ks * 32, C);
#pragma unroll
  for (int j = 0; j < 4; ++j) b[j] = load_frag_row(Brow + (size_t)j * 16 * C + ks * 32, C);
}

__device__ __forceinline__ void mma_step(const v16bf (&a)[2], const v16bf (&b)[4],
                                         v8f (&acc)[2][4]) {
#pragma unroll
  for (int i = 0; i < 2; ++i)
#pragma unroll
    for (int j = 0; j < 4; ++j)
      acc[i][j] = wmma_bf16(a[i], b[j], acc[i][j]);
}

// Double-buffered K-loop: issue k-step s+1 loads before k-step s WMMAs.
__device__ __forceinline__ void gemm_mainloop(const __bf16* Arow, const __bf16* Brow,
                                              v8f (&acc)[2][4]) {
  constexpr int KS = C / 32;               // 32
  v16bf a0[2], b0[4], a1[2], b1[4];
  load_ab(Arow, Brow, 0, a0, b0);
  for (int ks = 0; ks < KS - 2; ks += 2) {
    load_ab(Arow, Brow, ks + 1, a1, b1);
    mma_step(a0, b0, acc);
    load_ab(Arow, Brow, ks + 2, a0, b0);
    mma_step(a1, b1, acc);
  }
  load_ab(Arow, Brow, KS - 1, a1, b1);
  mma_step(a0, b0, acc);
  mma_step(a1, b1, acc);
}

// ---------------- GEMM 1: qkv = x @ Wqkv + b, scattered into q/k/v^T ----------------
__global__ __launch_bounds__(256) void k_gemm_qkv(
    const __bf16* __restrict__ A,   // [8192 x 1024]
    const __bf16* __restrict__ BT,  // [3072 x 1024]  (Wqkv^T)
    const float*  __restrict__ bias,// [3072]
    __bf16* __restrict__ qb,        // [B*H, T, Dh]  (pre-scaled by 1/8)
    __bf16* __restrict__ kb,        // [B*H, T, Dh]
    __bf16* __restrict__ vb) {      // [B*H, Dh, T]  (V transposed)
  const int wave = (blockIdx.x * blockDim.x + threadIdx.x) >> 5;
  const int lane = threadIdx.x & 31;
  const int NT4  = N_QKV / 64;                 // 48
  const int mt   = wave / NT4;                 // 0..255 (32-row tiles)
  const int nt   = wave % NT4;

  v8f acc[2][4] = { { {}, {}, {}, {} }, { {}, {}, {}, {} } };
  gemm_mainloop(A + (size_t)mt * 32 * C, BT + (size_t)nt * 64 * C, acc);

  const int col = lane & 15, hi = lane >> 4;
#pragma unroll
  for (int i = 0; i < 2; ++i) {
#pragma unroll
    for (int j = 0; j < 4; ++j) {
#pragma unroll
      for (int r = 0; r < 8; ++r) {
        int m = mt * 32 + i * 16 + r + 8 * hi;   // global row in [0, 8192)
        int n = nt * 64 + j * 16 + col;          // global col in [0, 3072)
        float v = acc[i][j][r] + bias[n];
        int b  = m >> 11;                        // / T
        int t  = m & (T - 1);
        int which = n >> 10;                     // 0=q 1=k 2=v
        int c  = n & (C - 1);
        int h  = c >> 6;
        int d  = c & (Dh - 1);
        int bh = b * H + h;
        if (which == 0)      qb[((size_t)bh * T + t) * Dh + d] = (__bf16)(v * 0.125f);
        else if (which == 1) kb[((size_t)bh * T + t) * Dh + d] = (__bf16)v;
        else                 vb[((size_t)bh * Dh + d) * T + t] = (__bf16)v;
      }
    }
  }
}

// ---------------- attention: one wave per (b,h, 32-query tile) ----------------
// Computes S^T = K Q^T (queries on lanes), online softmax, O^T = V^T P^T.
__global__ __launch_bounds__(256) void k_attn(
    const __bf16* __restrict__ qb,
    const __bf16* __restrict__ kb,
    const __bf16* __restrict__ vb,
    __bf16* __restrict__ attn) {     // [B*T, C] bf16
  const int wave = (blockIdx.x * blockDim.x + threadIdx.x) >> 5;
  const int lane = threadIdx.x & 31;
  const int qt = wave & (T / 32 - 1);          // 32-query tile, 0..63
  const int bh = wave >> 6;                    // 0..63

  const __bf16* q  = qb + (size_t)bh * T * Dh;
  const __bf16* kk = kb + (size_t)bh * T * Dh;
  const __bf16* vt = vb + (size_t)bh * Dh * T;

  // Q^T B-fragments: 2 query sub-tiles x 2 k-steps over Dh, loaded once
  v16bf bq[2][2];
#pragma unroll
  for (int u = 0; u < 2; ++u) {
    bq[u][0] = load_frag_row(q + (size_t)(qt * 32 + u * 16) * Dh + 0,  Dh);
    bq[u][1] = load_frag_row(q + (size_t)(qt * 32 + u * 16) * Dh + 32, Dh);
  }

  v8f o[2][4] = { { {}, {}, {}, {} }, { {}, {}, {}, {} } };
  float m_run[2] = { -1e30f, -1e30f };
  float l_run[2] = { 0.f, 0.f };

  auto load_k = [&](int k0, v16bf (&kf)[4]) {
    kf[0] = load_frag_row(kk + (size_t)k0 * Dh,             Dh);
    kf[1] = load_frag_row(kk + (size_t)k0 * Dh + 32,        Dh);
    kf[2] = load_frag_row(kk + (size_t)(k0 + 16) * Dh,      Dh);
    kf[3] = load_frag_row(kk + (size_t)(k0 + 16) * Dh + 32, Dh);
  };

  auto step = [&](int k0, const v16bf (&kf)[4]) {
    // scores for 32 keys x 32 queries: S^T tiles [q-sub][key-sub]
    v8f s[2][2];
#pragma unroll
    for (int u = 0; u < 2; ++u) {
      v8f z0 = {}, z1 = {};
      z0 = wmma_bf16(kf[0], bq[u][0], z0);
      z0 = wmma_bf16(kf[1], bq[u][1], z0);
      z1 = wmma_bf16(kf[2], bq[u][0], z1);
      z1 = wmma_bf16(kf[3], bq[u][1], z1);
      s[u][0] = z0; s[u][1] = z1;
    }
    // issue V loads now; softmax VALU below hides their latency
    v16bf vf[4];
#pragma unroll
    for (int jt = 0; jt < 4; ++jt)
      vf[jt] = load_frag_row(vt + (size_t)(jt * 16) * T + k0, T);

    // per-query (per-lane) online softmax; keys live in VGPRs + lane^16 partner
    float scal[2];
    v16bf pv[2];
#pragma unroll
    for (int u = 0; u < 2; ++u) {
      float mloc = -1e30f;
#pragma unroll
      for (int r = 0; r < 8; ++r) mloc = fmaxf(mloc, fmaxf(s[u][0][r], s[u][1][r]));
      mloc = fmaxf(mloc, __shfl_xor(mloc, 16, 32));
      const float m_new = fmaxf(m_run[u], mloc);
      const float sc = __expf(m_run[u] - m_new);
      float ls = 0.f;
      union { v16bf v; __bf16 e[16]; } up;
#pragma unroll
      for (int r = 0; r < 8; ++r) {
        float p0 = __expf(s[u][0][r] - m_new);
        float p1 = __expf(s[u][1][r] - m_new);
        ls += p0 + p1;
        up.e[r] = (__bf16)p0;            // key-within-32 = hi*8 + r
        up.e[r + 8] = (__bf16)p1;        // key-within-32 = 16 + hi*8 + r
      }
      l_run[u] = l_run[u] * sc + ls;     // partial; lane^16 holds the rest
      m_run[u] = m_new;
      scal[u] = sc; pv[u] = up.v;
    }

    // O^T accumulate: 4 d-tiles of 16, V fragment shared across both q sub-tiles
#pragma unroll
    for (int jt = 0; jt < 4; ++jt) {
#pragma unroll
      for (int u = 0; u < 2; ++u) {
        v8f oj = o[u][jt];
#pragma unroll
        for (int r = 0; r < 8; ++r) oj[r] *= scal[u];
        o[u][jt] = wmma_bf16(vf[jt], pv[u], oj);
      }
    }
  };

  // ping-pong K-fragment prefetch across 64 key tiles of 32
  v16bf kfA[4], kfB[4];
  load_k(0, kfA);
  for (int kt = 0; kt < T / 32; kt += 2) {
    load_k((kt + 1) * 32, kfB);          // prefetch odd tile
    step(kt * 32, kfA);
    if (kt + 2 < T / 32) load_k((kt + 2) * 32, kfA);  // prefetch next even tile
    step((kt + 1) * 32, kfB);
  }

  const int col = lane & 15, hi = lane >> 4;
  const int b = bh / H, h = bh % H;
#pragma unroll
  for (int u = 0; u < 2; ++u) {
    const float l_tot = l_run[u] + __shfl_xor(l_run[u], 16, 32);
    const float inv = 1.0f / l_tot;
    const int t = qt * 32 + u * 16 + col;
    const size_t rowbase = ((size_t)(b * T + t)) * C + h * Dh;
#pragma unroll
    for (int jt = 0; jt < 4; ++jt)
#pragma unroll
      for (int r = 0; r < 8; ++r)
        attn[rowbase + jt * 16 + r + 8 * hi] = (__bf16)(o[u][jt][r] * inv);
  }
}

// ---------------- GEMM 2: out = attn @ Wout + b (f32 output) ----------------
__global__ __launch_bounds__(256) void k_gemm_out(
    const __bf16* __restrict__ A,   // [8192 x 1024] attn bf16
    const __bf16* __restrict__ BT,  // [1024 x 1024] Wout^T bf16
    const float*  __restrict__ bias,// [1024]
    float* __restrict__ out) {      // [8192 x 1024] f32
  const int wave = (blockIdx.x * blockDim.x + threadIdx.x) >> 5;
  const int lane = threadIdx.x & 31;
  const int NT4  = C / 64;                     // 16
  const int mt   = wave / NT4;                 // 0..255
  const int nt   = wave % NT4;

  v8f acc[2][4] = { { {}, {}, {}, {} }, { {}, {}, {}, {} } };
  gemm_mainloop(A + (size_t)mt * 32 * C, BT + (size_t)nt * 64 * C, acc);

  const int col = lane & 15, hi = lane >> 4;
#pragma unroll
  for (int i = 0; i < 2; ++i) {
#pragma unroll
    for (int j = 0; j < 4; ++j) {
#pragma unroll
      for (int r = 0; r < 8; ++r) {
        int m = mt * 32 + i * 16 + r + 8 * hi;
        int n = nt * 64 + j * 16 + col;
        out[(size_t)m * C + n] = acc[i][j][r] + bias[n];
      }
    }
  }
}

// ---------------- launcher ----------------
extern "C" void kernel_launch(void* const* d_in, const int* in_sizes, int n_in,
                              void* d_out, int out_size, void* d_ws, size_t ws_size,
                              hipStream_t stream) {
  const float* x     = (const float*)d_in[0];
  const float* Wqkv  = (const float*)d_in[1];
  const float* bqkv  = (const float*)d_in[2];
  const float* Wout  = (const float*)d_in[3];
  const float* bout  = (const float*)d_in[4];

  char* ws = (char*)d_ws;
  __bf16* xb   = (__bf16*)(ws + OFF_XB);
  __bf16* wqT  = (__bf16*)(ws + OFF_WQT);
  __bf16* woT  = (__bf16*)(ws + OFF_WOT);
  __bf16* qb   = (__bf16*)(ws + OFF_QB);
  __bf16* kb   = (__bf16*)(ws + OFF_KB);
  __bf16* vb   = (__bf16*)(ws + OFF_VB);
  __bf16* attn = (__bf16*)(ws + OFF_ATT);

  // 1. convert / transpose (weights -> [N x K] bf16 so B fragments load row-major)
  {
    int n4 = (MTOT * C) / 4;                        // 2097152
    k_cvt_bf16<<<(n4 + 255) / 256, 256, 0, stream>>>(x, xb, n4);
  }
  k_transpose_bf16<<<(N_QKV / 32) * (C / 32), 256, 0, stream>>>(Wqkv, wqT, C, N_QKV);
  k_transpose_bf16<<<(C / 32) * (C / 32), 256, 0, stream>>>(Wout, woT, C, C);

  // 2. QKV projection: 256 m-tiles x 48 n-strips = 12288 waves, 8 waves/block
  k_gemm_qkv<<<(MTOT / 32) * (N_QKV / 64) / 8, 256, 0, stream>>>(xb, wqT, bqkv, qb, kb, vb);

  // 3. attention: 64 head*batch x 64 query tiles = 4096 waves
  k_attn<<<(BB * H) * (T / 32) / 8, 256, 0, stream>>>(qb, kb, vb, attn);

  // 4. output projection: 256 x 16 = 4096 waves
  k_gemm_out<<<(MTOT / 32) * (C / 64) / 8, 256, 0, stream>>>(attn, woT, bout, (float*)d_out);
}
